// MLSTM_30829275251146
// MI455X (gfx1250) — matrix operand order repaired and checked by reading, not compiled, not run on hardware
//
#include <hip/hip_runtime.h>
#include <hip/hip_bf16.h>

// ---------------------------------------------------------------------------
// mLSTM on MI455X (gfx1250):
//   Phase A (parallel): x, mx_w, W_ih converted once to bf16 (halves HBM
//     traffic: x is re-read 20x by the column blocks). P_xm / P_xg GEMMs are
//     double-buffered with GLOBAL_LOAD_ASYNC_TO_LDS_B128 (ASYNCcnt) staging
//     feeding v_wmma_f32_16x16x32_bf16, f32 accumulate.
//   Phase B (serial):   persistent single-workgroup kernel, 16 wave32s,
//     h (bf16) and m (bf16) resident in 256KB of the WGP's 320KB LDS,
//     recurrent weights bf16 in L2, 4-row-tile register blocking so each
//     weight B-fragment is reused 4x, c state in d_out (f32),
//     next-step activations prefetched into L2 (global_prefetch_b8).
// ---------------------------------------------------------------------------

#define T_DIM 1024
#define B_DIM 128
#define D_DIM 512
#define H_DIM 512

typedef __attribute__((ext_vector_type(16))) __bf16 v16bf;
typedef __attribute__((ext_vector_type(8)))  __bf16 v8bf;
typedef __attribute__((ext_vector_type(8)))  float  v8f;

// --- CDNA5 async global->LDS copy (16B), tracked by ASYNCcnt ---------------
__device__ inline void async_ld_b128(unsigned int lds_off, const void* gaddr) {
    asm volatile("global_load_async_to_lds_b128 %0, %1, off"
                 :: "v"(lds_off), "v"(gaddr)
                 : "memory");
}
__device__ inline void wait_async0() {
    asm volatile("s_wait_asynccnt 0x0" ::: "memory");
}
// LDS byte offset = low 32 bits of the generic (flat) LDS address.
__device__ inline unsigned int lds_off_of(const void* p) {
    return (unsigned int)(unsigned long long)p;
}

// A-matrix fragment (16x32 bf16), row-major source with leading dim `ld`.
// ISA 7.12.2: lane<16 holds row M=lane, K={0..7,16..23}; lane>=16 row M=lane-16,
// K={8..15,24..31}; packed pairs -> two contiguous 16B loads.
template <typename P>
__device__ inline v16bf frag_ld_a(P base, int ld, int lane) {
    const int m  = lane & 15;
    const int kb = (lane >> 4) << 3;     // 0 or 8
    P p = base + m * ld + kb;
    v8bf lo = *(const v8bf*)(p);         // K = kb .. kb+7
    v8bf hi = *(const v8bf*)(p + 16);    // K = kb+16 .. kb+23
    v16bf r;
#pragma unroll
    for (int i = 0; i < 8; ++i) { r[i] = lo[i]; r[i + 8] = hi[i]; }
    return r;
}

// B-matrix fragment (32x16 bf16) from row-major weights W[n][k] (N x K):
// lane<16 holds col N=lane, K={0..7,16..23}; lane>=16 col N=lane-16, K={8..15,24..31}.
template <typename P>
__device__ inline v16bf frag_ld_b(P base, int ld, int lane) {
    const int n  = lane & 15;
    const int kb = (lane >> 4) << 3;     // 0 or 8
    P p = base + n * ld + kb;
    v8bf lo = *(const v8bf*)(p);
    v8bf hi = *(const v8bf*)(p + 16);
    v16bf r;
#pragma unroll
    for (int i = 0; i < 8; ++i) { r[i] = lo[i]; r[i + 8] = hi[i]; }
    return r;
}

__device__ inline v8f wmma_bf16(v16bf a, v16bf b, v8f c) {
    return __builtin_amdgcn_wmma_f32_16x16x32_bf16(
        /*neg_a=*/false, a, /*neg_b=*/false, b,
        /*c_mod=*/(short)0, c, /*reuse_a=*/false, /*reuse_b=*/false);
}

__device__ inline float sigmoidf_(float x) { return 1.0f / (1.0f + __expf(-x)); }

// ---------------------------------------------------------------------------
__global__ void cvt_f32_bf16(const float* __restrict__ in,
                             __bf16* __restrict__ out, int n) {
    int i = blockIdx.x * blockDim.x + threadIdx.x;
    if (i < n) out[i] = (__bf16)in[i];
}

// ---------------------------------------------------------------------------
// Precompute GEMM (bf16 inputs): out[M,N] = bf16( X[M,D] @ W[N,D]^T + biases )
// Block = 256 threads (8 wave32s), 128x128 output tile, 32-wide K tiles.
// Double-buffered staging uses the async global->LDS DMA: issue next tile's
// async copies, run this tile's WMMAs, s_wait_asynccnt 0, barrier.
// ---------------------------------------------------------------------------
#define GLDA 48  // LDS leading dim (bf16); 96B rows keep 16B alignment
__global__ __launch_bounds__(256, 2)
void gemm_xw(const __bf16* __restrict__ X, const __bf16* __restrict__ W,
             const float* __restrict__ bias0, const float* __restrict__ bias1,
             __bf16* __restrict__ out, int M, int N, int D) {
    __shared__ __bf16 Abuf[2][128 * GLDA];   // 2 x 12 KB
    __shared__ __bf16 Bbuf[2][128 * GLDA];   // 2 x 12 KB

    const int tid  = threadIdx.x;
    const int wave = tid >> 5;
    const int lane = tid & 31;
    const int row0 = blockIdx.x * 128;
    const int col0 = blockIdx.y * 128;

    // staging coords: 128 rows x 4 16B-chunks (8 bf16) = 512 chunks, 2/thread
    int srow[2], skc[2];
#pragma unroll
    for (int it = 0; it < 2; ++it) {
        int c = tid * 2 + it;
        srow[it] = c >> 2;
        skc[it]  = (c & 3) * 8;
    }

    v8f acc[8];
#pragma unroll
    for (int i = 0; i < 8; ++i) acc[i] = (v8f){};

    // stage K-tile 0 into buffer 0 (async DMA), wait, barrier
#pragma unroll
    for (int it = 0; it < 2; ++it) {
        async_ld_b128(lds_off_of(&Abuf[0][srow[it] * GLDA + skc[it]]),
                      X + (size_t)(row0 + srow[it]) * D + skc[it]);
        async_ld_b128(lds_off_of(&Bbuf[0][srow[it] * GLDA + skc[it]]),
                      W + (size_t)(col0 + srow[it]) * D + skc[it]);
    }
    wait_async0();
    __syncthreads();

    int cur = 0;
    for (int k0 = 0; k0 < D; k0 += 32) {
        const bool has_next = (k0 + 32) < D;
        if (has_next) {
            const int nxt = cur ^ 1;
#pragma unroll
            for (int it = 0; it < 2; ++it) {
                async_ld_b128(lds_off_of(&Abuf[nxt][srow[it] * GLDA + skc[it]]),
                              X + (size_t)(row0 + srow[it]) * D + k0 + 32 + skc[it]);
                async_ld_b128(lds_off_of(&Bbuf[nxt][srow[it] * GLDA + skc[it]]),
                              W + (size_t)(col0 + srow[it]) * D + k0 + 32 + skc[it]);
            }
            if (k0 + 64 < D) {  // warm L2 for the K+2 tile -> global_prefetch_b8
                __builtin_prefetch(X + (size_t)(row0 + srow[0]) * D + k0 + 64 + skc[0], 0, 1);
                __builtin_prefetch(W + (size_t)(col0 + srow[0]) * D + k0 + 64 + skc[0], 0, 1);
            }
        }

        // compute on current buffer while the async DMA fills the other one
        v16bf af = frag_ld_a(&Abuf[cur][(wave * 16) * GLDA], GLDA, lane);
#pragma unroll
        for (int ct = 0; ct < 8; ++ct) {
            v16bf bf = frag_ld_b(&Bbuf[cur][(ct * 16) * GLDA], GLDA, lane);
            acc[ct] = wmma_bf16(af, bf, acc[ct]);
        }

        if (has_next) wait_async0();   // my async writes to nxt are in LDS
        __syncthreads();               // everyone's writes visible
        cur ^= 1;
    }

    // Epilogue: C layout (M = v + 8*(lane>=16), N = lane&15)
    const int rbase = row0 + wave * 16 + ((lane >> 4) << 3);
#pragma unroll
    for (int ct = 0; ct < 8; ++ct) {
        int col = col0 + ct * 16 + (lane & 15);
        float b = bias0[col] + (bias1 ? bias1[col] : 0.0f);
#pragma unroll
        for (int v = 0; v < 8; ++v) {
            out[(size_t)(rbase + v) * N + col] = (__bf16)(acc[ct][v] + b);
        }
    }
}

// ---------------------------------------------------------------------------
// Serial recurrence: one workgroup, 512 threads = 16 wave32s, persistent over
// all T steps. h and m live in LDS (bf16, 128KB each); c lives in d_out (f32).
// 4-row-tile register blocking: each weight B-fragment is loaded once and
// used by 4 WMMAs (4x less L2 weight traffic per step).
// ---------------------------------------------------------------------------
__global__ __launch_bounds__(512, 1)
void mlstm_recurrent(const __bf16* __restrict__ Pxm,   // [T][B][H]
                     const __bf16* __restrict__ Pxg,   // [T][B][4H]
                     const __bf16* __restrict__ mhw,   // [H][H]  bf16
                     const __bf16* __restrict__ whh,   // [4H][H] bf16
                     const float* __restrict__ mh_b,   // [H]
                     const int*   __restrict__ lengths,// [B]
                     const float* __restrict__ h0,
                     const float* __restrict__ c0,
                     float* __restrict__ c_out)        // [B][H] state + output
{
    extern __shared__ __bf16 smem[];
    __bf16* hbuf = smem;                     // [B][H] bf16
    __bf16* mbuf = smem + B_DIM * H_DIM;     // [B][H] bf16

    const int tid  = threadIdx.x;
    const int wave = tid >> 5;               // 0..15
    const int lane = tid & 31;

    for (int i = tid; i < B_DIM * H_DIM; i += 512) {
        int n = i & (H_DIM - 1);
        hbuf[i]  = (__bf16)h0[n];
        c_out[i] = c0[n];
    }
    __syncthreads();

    for (int t = 0; t < T_DIM; ++t) {
        const __bf16* xm_t = Pxm + (size_t)t * B_DIM * H_DIM;
        const __bf16* xg_t = Pxg + (size_t)t * B_DIM * 4 * H_DIM;

        // Warm L2 for next step's activations (global_prefetch_b8).
        if (t + 1 < T_DIM) {
            const char* nxm = (const char*)(xm_t + B_DIM * H_DIM);
            const char* nxg = (const char*)(xg_t + B_DIM * 4 * H_DIM);
            __builtin_prefetch(nxm + tid * 256, 0, 1);          // 128 KB
#pragma unroll
            for (int q = 0; q < 4; ++q)                         // 512 KB
                __builtin_prefetch(nxg + (q * 512 + tid) * 256, 0, 1);
        }

        // ---- Phase 1: hm = h @ mh_w^T + mh_b ; m = xm * hm --------------
        // items: 2 row-blocks (4 row-tiles each) x 32 col-tiles = 64, 4/wave
        for (int item = 0; item < 4; ++item) {
            const int id = wave * 4 + item;      // 0..63
            const int rb = id >> 5;              // 0..1
            const int nt = id & 31;              // 0..31
            v8f acc[4];
#pragma unroll
            for (int rr = 0; rr < 4; ++rr) acc[rr] = (v8f){};
            for (int k0 = 0; k0 < H_DIM; k0 += 32) {
                v16bf bf = frag_ld_b(mhw + (size_t)(nt * 16) * H_DIM + k0, H_DIM, lane);
#pragma unroll
                for (int rr = 0; rr < 4; ++rr) {
                    v16bf af = frag_ld_a(hbuf + ((rb * 4 + rr) * 16) * H_DIM + k0, H_DIM, lane);
                    acc[rr] = wmma_bf16(af, bf, acc[rr]);
                }
            }
            const int col  = nt * 16 + (lane & 15);
            const float bias = mh_b[col];
#pragma unroll
            for (int rr = 0; rr < 4; ++rr) {
                const int rbase = (rb * 4 + rr) * 16 + ((lane >> 4) << 3);
#pragma unroll
                for (int v = 0; v < 8; ++v) {
                    int row = rbase + v;
                    float hm = acc[rr][v] + bias;
                    float xm = (float)xm_t[(size_t)row * H_DIM + col];
                    mbuf[row * H_DIM + col] = (__bf16)(xm * hm);
                }
            }
        }
        __syncthreads();  // m complete; all h reads for this step done

        // ---- Phase 2: gates = P_xg[t] + m @ W_hh^T ; masked update ------
        for (int item = 0; item < 4; ++item) {
            const int id = wave * 4 + item;
            const int rb = id >> 5;
            const int nt = id & 31;
            v8f acc[4][4];   // [gate][row-tile]
#pragma unroll
            for (int g = 0; g < 4; ++g)
#pragma unroll
                for (int rr = 0; rr < 4; ++rr) acc[g][rr] = (v8f){};

            for (int k0 = 0; k0 < H_DIM; k0 += 32) {
                v16bf am[4];
#pragma unroll
                for (int rr = 0; rr < 4; ++rr)
                    am[rr] = frag_ld_a(mbuf + ((rb * 4 + rr) * 16) * H_DIM + k0, H_DIM, lane);
#pragma unroll
                for (int g = 0; g < 4; ++g) {
                    v16bf bg = frag_ld_b(whh + (size_t)(g * H_DIM + nt * 16) * H_DIM + k0,
                                         H_DIM, lane);
#pragma unroll
                    for (int rr = 0; rr < 4; ++rr)
                        acc[g][rr] = wmma_bf16(am[rr], bg, acc[g][rr]);
                }
            }

            const int col = nt * 16 + (lane & 15);
#pragma unroll
            for (int rr = 0; rr < 4; ++rr) {
                const int rbase = (rb * 4 + rr) * 16 + ((lane >> 4) << 3);
#pragma unroll
                for (int v = 0; v < 8; ++v) {
                    const int row = rbase + v;
                    const size_t gb = (size_t)row * 4 * H_DIM;
                    float gi = sigmoidf_(acc[0][rr][v] + (float)xg_t[gb + 0 * H_DIM + col]);
                    float gf = sigmoidf_(acc[1][rr][v] + (float)xg_t[gb + 1 * H_DIM + col]);
                    float gg = tanhf(acc[2][rr][v]     + (float)xg_t[gb + 2 * H_DIM + col]);
                    float go = sigmoidf_(acc[3][rr][v] + (float)xg_t[gb + 3 * H_DIM + col]);
                    const int idx = row * H_DIM + col;
                    float c_old = c_out[idx];
                    float c_new = gf * c_old + gi * gg;
                    bool  msk   = t < lengths[row];
                    c_out[idx]  = msk ? c_new : c_old;
                    float h_old = (float)hbuf[idx];
                    float h_new = msk ? go * tanhf(c_new) : h_old;
                    hbuf[idx]   = (__bf16)h_new;
                }
            }
        }
        __syncthreads();  // h_new visible for next step's phase 1
    }
}

// ---------------------------------------------------------------------------
extern "C" void kernel_launch(void* const* d_in, const int* in_sizes, int n_in,
                              void* d_out, int out_size, void* d_ws, size_t ws_size,
                              hipStream_t stream) {
    const float* x      = (const float*)d_in[0];   // [T,B,D]
    const int*   lens   = (const int*)  d_in[1];   // [B]
    const float* W_ih   = (const float*)d_in[2];   // [4H,D]
    const float* b_ih   = (const float*)d_in[3];   // [4H]
    const float* W_hh   = (const float*)d_in[4];   // [4H,H]
    const float* b_hh   = (const float*)d_in[5];   // [4H]
    const float* mx_w   = (const float*)d_in[6];   // [H,D]
    const float* mx_b   = (const float*)d_in[7];   // [H]
    const float* mh_w   = (const float*)d_in[8];   // [H,H]
    const float* mh_b   = (const float*)d_in[9];   // [H]
    const float* h0     = (const float*)d_in[10];  // [H]
    const float* c0     = (const float*)d_in[11];  // [H]
    float* c_out = (float*)d_out;                  // [B,H]

    // Workspace layout
    char* ws = (char*)d_ws;
    size_t off = 0;
    __bf16* Pxm    = (__bf16*)(ws + off); off += (size_t)T_DIM * B_DIM * H_DIM * 2;      // 128 MB
    __bf16* Pxg    = (__bf16*)(ws + off); off += (size_t)T_DIM * B_DIM * 4 * H_DIM * 2;  // 512 MB
    __bf16* mhw_bf = (__bf16*)(ws + off); off += (size_t)H_DIM * H_DIM * 2;              // 512 KB
    __bf16* whh_bf = (__bf16*)(ws + off); off += (size_t)4 * H_DIM * H_DIM * 2;          // 2 MB
    __bf16* mxw_bf = (__bf16*)(ws + off); off += (size_t)H_DIM * D_DIM * 2;              // 512 KB
    __bf16* wih_bf = (__bf16*)(ws + off); off += (size_t)4 * H_DIM * D_DIM * 2;          // 2 MB
    __bf16* Xbf    = (__bf16*)(ws + off);                                                // 128 MB

    // 1) one-time f32 -> bf16 conversions (x + all GEMM weights)
    {
        int n;
        n = H_DIM * H_DIM;
        cvt_f32_bf16<<<(n + 255) / 256, 256, 0, stream>>>(mh_w, mhw_bf, n);
        n = 4 * H_DIM * H_DIM;
        cvt_f32_bf16<<<(n + 255) / 256, 256, 0, stream>>>(W_hh, whh_bf, n);
        n = H_DIM * D_DIM;
        cvt_f32_bf16<<<(n + 255) / 256, 256, 0, stream>>>(mx_w, mxw_bf, n);
        n = 4 * H_DIM * D_DIM;
        cvt_f32_bf16<<<(n + 255) / 256, 256, 0, stream>>>(W_ih, wih_bf, n);
        n = T_DIM * B_DIM * D_DIM;  // 67,108,864
        cvt_f32_bf16<<<(n + 255) / 256, 256, 0, stream>>>(x, Xbf, n);
    }

    // 2) time-parallel GEMMs (x-side, biases folded), async-LDS staging
    {
        const int M = T_DIM * B_DIM;  // 131072
        dim3 g1(M / 128, H_DIM / 128);
        gemm_xw<<<g1, 256, 0, stream>>>(Xbf, mxw_bf, mx_b, nullptr, Pxm, M, H_DIM, D_DIM);
        dim3 g2(M / 128, (4 * H_DIM) / 128);
        gemm_xw<<<g2, 256, 0, stream>>>(Xbf, wih_bf, b_ih, b_hh, Pxg, M, 4 * H_DIM, D_DIM);
    }

    // 3) serial recurrence: single WGP, 256 KB dynamic LDS (h + m, bf16)
    {
        const size_t lds = (size_t)2 * B_DIM * H_DIM * sizeof(__bf16);  // 256 KB
        hipFuncSetAttribute((const void*)mlstm_recurrent,
                            hipFuncAttributeMaxDynamicSharedMemorySize, (int)lds);
        mlstm_recurrent<<<1, 512, lds, stream>>>(Pxm, Pxg, mhw_bf, whh_bf,
                                                 mh_b, lens, h0, c0, c_out);
    }
}
